// MIX_OD_ATT_87522843558435
// MI455X (gfx1250) — matrix operand order repaired and checked by reading, not compiled, not run on hardware
//
#include <hip/hip_runtime.h>

// MI455X / gfx1250, wave32, WMMA f16->f32 (V_WMMA_F32_16X16X32_F16).
// All GEMM operands stored in f16 (K padded to multiple of 32, weights
// pre-transposed to [N][Kp]) so fragment loads are raw 16/32-byte vector
// loads with zero in-loop conversions. f32 only where math requires it.

typedef __attribute__((ext_vector_type(16))) _Float16 v16h;
typedef __attribute__((ext_vector_type(8)))  _Float16 v8h;
typedef __attribute__((ext_vector_type(8)))  float    v8f;

#define BT    2048   // B*T tokens
#define TT    128    // T
#define HEADS 8
#define DHD   32     // per-head dim
#define FIN   193
#define FINP  224    // FIN padded to multiple of 32
#define HID   256
#define OHID  288    // 32 + 256 (already multiple of 32)

// float4 x4 -> v16h (only used where source is f32 LDS/registers)
__device__ inline v16h cvt16(float4 a0, float4 a1, float4 a2, float4 a3) {
  v16h r;
  r[0]=(_Float16)a0.x;  r[1]=(_Float16)a0.y;  r[2]=(_Float16)a0.z;  r[3]=(_Float16)a0.w;
  r[4]=(_Float16)a1.x;  r[5]=(_Float16)a1.y;  r[6]=(_Float16)a1.z;  r[7]=(_Float16)a1.w;
  r[8]=(_Float16)a2.x;  r[9]=(_Float16)a2.y;  r[10]=(_Float16)a2.z; r[11]=(_Float16)a2.w;
  r[12]=(_Float16)a3.x; r[13]=(_Float16)a3.y; r[14]=(_Float16)a3.z; r[15]=(_Float16)a3.w;
  return r;
}

// two contiguous 8-half runs -> A fragment
__device__ inline v16h packA(v8h lo, v8h hi) {
  v16h r;
#pragma unroll
  for (int i = 0; i < 8; ++i) { r[i] = lo[i]; r[8 + i] = hi[i]; }
  return r;
}

// ---------------------------------------------------------------------------
// Weight prep: WT[z][n][kp] = f16((kp < K) ? W[z][kp][n] : 0)
// ---------------------------------------------------------------------------
__global__ __launch_bounds__(256)
void transpose_pad(const float* __restrict__ W, _Float16* __restrict__ WT,
                   int K, int N, int Kp, long wStrideZ, long wtStrideZ)
{
  int z = blockIdx.z;
  long idx = (long)blockIdx.x * 256 + threadIdx.x;
  long tot = (long)N * Kp;
  if (idx >= tot) return;
  int n  = (int)(idx / Kp);
  int kp = (int)(idx % Kp);
  float v = (kp < K) ? W[(long)z * wStrideZ + (long)kp * N + n] : 0.0f;
  WT[(long)z * wtStrideZ + idx] = (_Float16)v;
}

// ---------------------------------------------------------------------------
// WMMA GEMM: act(A[M,Kp] @ WT[z][N,Kp]^T + bias[z][N]) -> Cf (f32) / Ch (f16)
// A, WT in f16; one wave per block; NT adjacent 16x16 N-tiles per wave with
// A-fragment reuse. act: 0=none, 1=exact GELU.
// ---------------------------------------------------------------------------
template<int NT>
__global__ __launch_bounds__(32)
void gemm_wmma(const _Float16* __restrict__ A, const _Float16* __restrict__ WT,
               const float* __restrict__ bias,
               float* __restrict__ Cf, _Float16* __restrict__ Ch,
               int N, int Kp,
               long aStrideZ, long wStrideZ, long bStrideZ, long cStrideZ,
               int act)
{
  int lane = threadIdx.x;
  int nt0 = blockIdx.x * NT, mt = blockIdx.y, z = blockIdx.z;
  const _Float16* Az = A  + (long)z * aStrideZ;
  const _Float16* Wz = WT + (long)z * wStrideZ;
  const float*    Bz = bias ? (bias + (long)z * bStrideZ) : nullptr;

  int grp = lane >> 4, l15 = lane & 15;
  const _Float16* Ar = Az + (long)(mt * 16 + l15) * Kp + grp * 16;
  const _Float16* Br[NT];
#pragma unroll
  for (int t = 0; t < NT; ++t)
    Br[t] = Wz + (long)((nt0 + t) * 16 + l15) * Kp + grp * 16;

  v8f acc[NT];
#pragma unroll
  for (int t = 0; t < NT; ++t) acc[t] = (v8f){};

  for (int k0 = 0; k0 < Kp; k0 += 32) {
    v16h a = packA(*(const v8h*)(Ar + k0), *(const v8h*)(Ar + k0 + 8));
#pragma unroll
    for (int t = 0; t < NT; ++t) {
      v16h b = *(const v16h*)(Br[t] + k0);          // 32B contiguous
      acc[t] = __builtin_amdgcn_wmma_f32_16x16x32_f16(false, a, false, b,
                                                      (short)0, acc[t], false, false);
    }
  }
#pragma unroll
  for (int t = 0; t < NT; ++t) {
    int col = (nt0 + t) * 16 + l15;
    float bval = Bz ? Bz[col] : 0.0f;
#pragma unroll
    for (int r = 0; r < 8; ++r) {
      int row = mt * 16 + r + 8 * grp;
      float vv = acc[t][r] + bval;
      if (act == 1) vv = 0.5f * vv * (1.0f + erff(vv * 0.70710678118654752f));
      long idx = (long)z * cStrideZ + (long)row * N + col;
      if (Cf) Cf[idx] = vv;
      if (Ch) Ch[idx] = (_Float16)vv;
    }
  }
}

// ---------------------------------------------------------------------------
// Embedding into padded f16 x: [tau|Wo+bo|Wd+bd|Wt+bt|Ww+bw|zeros]
// ---------------------------------------------------------------------------
__global__ __launch_bounds__(64)
void embed_kernel(const float* __restrict__ tau,
                  const int* __restrict__ oi, const int* __restrict__ di,
                  const int* __restrict__ ti, const int* __restrict__ wi,
                  const float* __restrict__ Wo, const float* __restrict__ bo,
                  const float* __restrict__ Wd, const float* __restrict__ bd,
                  const float* __restrict__ Wt, const float* __restrict__ bt,
                  const float* __restrict__ Ww, const float* __restrict__ bw,
                  _Float16* __restrict__ x)
{
  int m = blockIdx.x, tid = threadIdx.x;
  int io = oi[m], id = di[m], it = ti[m], iw = wi[m];
  _Float16* xr = x + (long)m * FINP;
  for (int c = tid; c < FINP; c += 64) {
    float val = 0.0f;
    if      (c == 0)  val = tau[m];
    else if (c < 65)  val = Wo[io * 64 + (c - 1)]   + bo[c - 1];
    else if (c < 129) val = Wd[id * 64 + (c - 65)]  + bd[c - 65];
    else if (c < 161) val = Wt[it * 32 + (c - 129)] + bt[c - 129];
    else if (c < FIN) val = Ww[iw * 32 + (c - 161)] + bw[c - 161];
    xr[c] = (_Float16)val;
  }
}

// ---------------------------------------------------------------------------
// Causal attention per (b,h). q/k/v layout: [H][B*T][32] f16.
// WMMA QK^T -> 64KB LDS f32 scores -> causal softmax -> WMMA P@V.
// attn_x output: [B*T][H*32] f16 (token-major, heads concatenated).
// ---------------------------------------------------------------------------
__global__ __launch_bounds__(256)
void attn_kernel(const _Float16* __restrict__ q, const _Float16* __restrict__ k,
                 const _Float16* __restrict__ v, _Float16* __restrict__ attn_x)
{
  __shared__ float sc[TT * TT];                 // 64 KB
  int tid = threadIdx.x;
  int lane = tid & 31, w = tid >> 5;            // 8 waves
  int b = blockIdx.x / HEADS, h = blockIdx.x % HEADS;
  long base = ((long)h * BT + (long)b * TT) * DHD;
  int grp = lane >> 4, l15 = lane & 15;
  int mt = w;                                   // each wave: 16-row strip

  // ---- scores = (q @ k^T) / sqrt(32), K = 32 (one WMMA chunk) ----
  const _Float16* qr = q + base + (long)(mt * 16 + l15) * DHD + grp * 16;
  v16h aq = packA(*(const v8h*)(qr), *(const v8h*)(qr + 8));
  for (int nt = 0; nt < 8; ++nt) {
    const _Float16* kr = k + base + (long)(nt * 16 + l15) * DHD + grp * 16;
    v16h bk = *(const v16h*)(kr);
    v8f s = {};
    s = __builtin_amdgcn_wmma_f32_16x16x32_f16(false, aq, false, bk,
                                               (short)0, s, false, false);
#pragma unroll
    for (int r = 0; r < 8; ++r)
      sc[(mt * 16 + r + 8 * grp) * TT + nt * 16 + l15] = s[r] * 0.17677669529663687f;
  }
  __syncthreads();

  // ---- causal softmax, one thread per row ----
  if (tid < TT) {
    int r = tid;
    float mx = -1e30f;
    for (int s2 = 0; s2 <= r; ++s2) mx = fmaxf(mx, sc[r * TT + s2]);
    float sum = 0.0f;
    for (int s2 = 0; s2 <= r; ++s2) {
      float e = expf(sc[r * TT + s2] - mx);
      sc[r * TT + s2] = e; sum += e;
    }
    float inv = 1.0f / sum;
    for (int s2 = 0; s2 <= r; ++s2) sc[r * TT + s2] *= inv;
    for (int s2 = r + 1; s2 < TT; ++s2) sc[r * TT + s2] = 0.0f;
  }
  __syncthreads();

  // ---- attn = P @ V, K = 128 (4 WMMA chunks), N = 32 (2 tiles) ----
  const float* pr = &sc[(long)(mt * 16 + l15) * TT + grp * 8];
  for (int nt = 0; nt < 2; ++nt) {
    v8f acc = {};
    for (int k0 = 0; k0 < TT; k0 += 32) {
      v16h ap = cvt16(*(const float4*)(pr + k0),      *(const float4*)(pr + k0 + 4),
                      *(const float4*)(pr + k0 + 16), *(const float4*)(pr + k0 + 20));
      v16h bvv;
#pragma unroll
      for (int i = 0; i < 16; ++i) {
        int ks2 = k0 + ((i < 8) ? i : (i + 8)) + grp * 8;   // strided over tokens, unconditional
        bvv[i] = v[base + (long)ks2 * DHD + nt * 16 + l15];
      }
      acc = __builtin_amdgcn_wmma_f32_16x16x32_f16(false, ap, false, bvv,
                                                   (short)0, acc, false, false);
    }
#pragma unroll
    for (int r = 0; r < 8; ++r) {
      int t = mt * 16 + r + 8 * grp;
      int d = nt * 16 + l15;
      attn_x[((long)b * TT + t) * HID + h * DHD + d] = (_Float16)acc[r];
    }
  }
}

// ---------------------------------------------------------------------------
// time-mix parameters + build f16 o_hidden = [time_para(32) | hidden(256)]
// ---------------------------------------------------------------------------
__global__ __launch_bounds__(64)
void timemix_kernel(const float* __restrict__ raw, const _Float16* __restrict__ hiddenh,
                    _Float16* __restrict__ ohid)
{
  int m = blockIdx.x, tid = threadIdx.x;
  if (tid == 0) {
    const float* r = raw + (long)m * 32;
    _Float16* o = ohid + (long)m * OHID;
    float mx = -1e30f;
    for (int i = 0; i < 8; ++i) mx = fmaxf(mx, r[i]);
    float e[8], s = 0.0f;
    for (int i = 0; i < 8; ++i) { e[i] = expf(r[i] - mx); s += e[i]; }
    float inv = 1.0f / s;
    for (int i = 0; i < 8; ++i)  o[i] = (_Float16)(e[i] * inv);   // softmax(w)
    for (int i = 8; i < 16; ++i) o[i] = (_Float16)r[i];           // mu
    for (int i = 16; i < 32; ++i) {                               // softplus
      float x = r[i];
      o[i] = (_Float16)((x > 20.0f) ? x : log1pf(expf(x)));
    }
  }
  for (int j = tid; j < HID; j += 64)
    ohid[(long)m * OHID + 32 + j] = hiddenh[(long)m * HID + j];
}

// ---------------------------------------------------------------------------
// In-place softmax over 256 classes (one block per token)
// ---------------------------------------------------------------------------
__global__ __launch_bounds__(256)
void softmax256_kernel(float* __restrict__ x)
{
  __shared__ float red[256];
  int m = blockIdx.x, tid = threadIdx.x;
  float v = x[(long)m * 256 + tid];
  red[tid] = v; __syncthreads();
  for (int s = 128; s > 0; s >>= 1) {
    if (tid < s) red[tid] = fmaxf(red[tid], red[tid + s]);
    __syncthreads();
  }
  float mx = red[0]; __syncthreads();
  float e = expf(v - mx);
  red[tid] = e; __syncthreads();
  for (int s = 128; s > 0; s >>= 1) {
    if (tid < s) red[tid] += red[tid + s];
    __syncthreads();
  }
  x[(long)m * 256 + tid] = e / red[0];
}

// ---------------------------------------------------------------------------
// Fused OD head, one block per token (d1/d2 already f16 in global):
//   od = d1(256x16) @ d2^T, diag -> -1e9, row-softmax, d_out = o_out @ od_att.
// Never materializes the 537MB [B,T,256,256] tensor: 16-row strips in LDS.
// ---------------------------------------------------------------------------
__global__ __launch_bounds__(256)
void od_kernel(const _Float16* __restrict__ d1, const _Float16* __restrict__ d2,
               const float* __restrict__ oprob, float* __restrict__ out)
{
  __shared__ _Float16 d1h[256][18];
  __shared__ _Float16 d2h[256][18];
  __shared__ float sstrip[16][257];
  __shared__ float opl[256];
  __shared__ float red[16][16];
  __shared__ float rowinv[16];

  int m = blockIdx.x, tid = threadIdx.x;
  int lane = tid & 31, w = tid >> 5;
  int grp = lane >> 4, l15 = lane & 15;

  const _Float16* d1p = d1 + (long)m * 4096;
  const _Float16* d2p = d2 + (long)m * 4096;
  for (int j8 = tid; j8 < 512; j8 += 256) {       // 512 chunks of 8 halves
    int r = j8 >> 1, c = (j8 & 1) * 8;
    v8h t1 = *(const v8h*)(d1p + j8 * 8);
    v8h t2 = *(const v8h*)(d2p + j8 * 8);
#pragma unroll
    for (int e = 0; e < 8; ++e) { d1h[r][c + e] = t1[e]; d2h[r][c + e] = t2[e]; }
  }
  opl[tid] = oprob[(long)m * 256 + tid];
  float acc = 0.0f;
  __syncthreads();

  for (int strip = 0; strip < 16; ++strip) {
    // 16 col-tiles across 8 waves (2 each); K=16 zero-padded to 32
    for (int t = 0; t < 2; ++t) {
      int nt = w * 2 + t;
      v16h a, bfr;
#pragma unroll
      for (int i = 0; i < 16; ++i) {
        int ka = i + grp * 16;
        a[i]   = (ka < 16) ? d1h[strip * 16 + l15][ka] : (_Float16)0.0f;
        int kb = i + grp * 16;
        bfr[i] = (kb < 16) ? d2h[nt * 16 + l15][kb]   : (_Float16)0.0f;
      }
      v8f c = {};
      c = __builtin_amdgcn_wmma_f32_16x16x32_f16(false, a, false, bfr,
                                                 (short)0, c, false, false);
#pragma unroll
      for (int r = 0; r < 8; ++r)
        sstrip[r + 8 * grp][nt * 16 + l15] = c[r];
    }
    __syncthreads();
    if (tid < 16) sstrip[tid][strip * 16 + tid] = -1e9f;   // own_mask diagonal
    __syncthreads();

    // row softmax: 16 threads per row, 16 cols each
    int rr = tid >> 4, jj = tid & 15;
    float pm = -1e30f;
    for (int e2 = jj * 16; e2 < jj * 16 + 16; ++e2) pm = fmaxf(pm, sstrip[rr][e2]);
    red[rr][jj] = pm;
    __syncthreads();
    float mx = -1e30f;
    for (int j = 0; j < 16; ++j) mx = fmaxf(mx, red[rr][j]);
    __syncthreads();
    float ps = 0.0f;
    for (int e2 = jj * 16; e2 < jj * 16 + 16; ++e2) {
      float ev = expf(sstrip[rr][e2] - mx);
      sstrip[rr][e2] = ev; ps += ev;
    }
    red[rr][jj] = ps;
    __syncthreads();
    if (jj == 0) {
      float s = 0.0f;
      for (int j = 0; j < 16; ++j) s += red[rr][j];
      rowinv[rr] = 1.0f / s;
    }
    __syncthreads();

    // acc[e=tid] += sum_r o_out[f] * softmax_row_f[e]
    for (int r = 0; r < 16; ++r)
      acc += opl[strip * 16 + r] * sstrip[r][tid] * rowinv[r];
    __syncthreads();
  }
  out[(long)m * 256 + tid] = acc;
}

// ---------------------------------------------------------------------------
extern "C" void kernel_launch(void* const* d_in, const int* in_sizes, int n_in,
                              void* d_out, int out_size, void* d_ws, size_t ws_size,
                              hipStream_t stream)
{
  (void)in_sizes; (void)n_in; (void)out_size; (void)ws_size;
  const float* tau = (const float*)d_in[0];
  const int*   oi  = (const int*)d_in[1];
  const int*   di  = (const int*)d_in[2];
  const int*   ti  = (const int*)d_in[3];
  const int*   wi  = (const int*)d_in[4];
  const float* Wo  = (const float*)d_in[5];  const float* bo  = (const float*)d_in[6];
  const float* Wd  = (const float*)d_in[7];  const float* bd  = (const float*)d_in[8];
  const float* Wt  = (const float*)d_in[9];  const float* bt  = (const float*)d_in[10];
  const float* Ww  = (const float*)d_in[11]; const float* bw  = (const float*)d_in[12];
  const float* Wq  = (const float*)d_in[13]; const float* bq  = (const float*)d_in[14];
  const float* Wk  = (const float*)d_in[15]; const float* bk  = (const float*)d_in[16];
  const float* Wv  = (const float*)d_in[17]; const float* bv  = (const float*)d_in[18];
  const float* Watt= (const float*)d_in[19];
  const float* Wtm = (const float*)d_in[20]; const float* btm = (const float*)d_in[21];
  const float* Wol = (const float*)d_in[22]; const float* bol = (const float*)d_in[23];
  const float* Wd1 = (const float*)d_in[24]; const float* bd1 = (const float*)d_in[25];
  const float* Wd2 = (const float*)d_in[26]; const float* bd2 = (const float*)d_in[27];

  // workspace carve-up (bytes); total ~46MB, all blocks 16B-aligned
  char* p = (char*)d_ws;
  _Float16* xp     = (_Float16*)p; p += (long)BT * FINP * 2;        // 917504
  _Float16* qbuf   = (_Float16*)p; p += (long)HEADS * BT * DHD * 2; // 1048576
  _Float16* kbuf   = (_Float16*)p; p += (long)HEADS * BT * DHD * 2;
  _Float16* vbuf   = (_Float16*)p; p += (long)HEADS * BT * DHD * 2;
  _Float16* attnx  = (_Float16*)p; p += (long)BT * HID * 2;         // 1048576
  _Float16* hiddenh= (_Float16*)p; p += (long)BT * HID * 2;
  float*    rawb   = (float*)p;    p += (long)BT * 32 * 4;          // 262144
  _Float16* ohidh  = (_Float16*)p; p += (long)BT * OHID * 2;        // 1179648
  _Float16* d1b    = (_Float16*)p; p += (long)BT * 4096 * 2;        // 16777216
  _Float16* d2b    = (_Float16*)p; p += (long)BT * 4096 * 2;
  _Float16* Wqt    = (_Float16*)p; p += (long)HEADS * 32 * FINP * 2; // 114688
  _Float16* Wkt    = (_Float16*)p; p += (long)HEADS * 32 * FINP * 2;
  _Float16* Wvt    = (_Float16*)p; p += (long)HEADS * 32 * FINP * 2;
  _Float16* Wattt  = (_Float16*)p; p += (long)256 * 256 * 2;
  _Float16* Wtmt   = (_Float16*)p; p += (long)32 * 256 * 2;
  _Float16* Wolt   = (_Float16*)p; p += (long)256 * OHID * 2;
  _Float16* Wd1t   = (_Float16*)p; p += (long)4096 * OHID * 2;
  _Float16* Wd2t   = (_Float16*)p; p += (long)4096 * OHID * 2;

  float* hidden = (float*)d_out;            // output 0: [2048,256] f32
  float* oout   = hidden + 524288;          // output 1: [2048,256] f32
  float* dfin   = oout   + 524288;          // output 2: [2048,256] f32

  // 0) weight prep: transpose + K-pad + f16 (one-time, ~5MB f16)
  transpose_pad<<<dim3(28, 1, 8), 256, 0, stream>>>(Wq, Wqt, FIN, 32, FINP,
                                                    (long)FIN * 32, (long)32 * FINP);
  transpose_pad<<<dim3(28, 1, 8), 256, 0, stream>>>(Wk, Wkt, FIN, 32, FINP,
                                                    (long)FIN * 32, (long)32 * FINP);
  transpose_pad<<<dim3(28, 1, 8), 256, 0, stream>>>(Wv, Wvt, FIN, 32, FINP,
                                                    (long)FIN * 32, (long)32 * FINP);
  transpose_pad<<<dim3(256, 1, 1), 256, 0, stream>>>(Watt, Wattt, 256, 256, 256, 0, 0);
  transpose_pad<<<dim3(32, 1, 1), 256, 0, stream>>>(Wtm, Wtmt, 256, 32, 256, 0, 0);
  transpose_pad<<<dim3(288, 1, 1), 256, 0, stream>>>(Wol, Wolt, OHID, 256, OHID, 0, 0);
  transpose_pad<<<dim3(4608, 1, 1), 256, 0, stream>>>(Wd1, Wd1t, OHID, 4096, OHID, 0, 0);
  transpose_pad<<<dim3(4608, 1, 1), 256, 0, stream>>>(Wd2, Wd2t, OHID, 4096, OHID, 0, 0);

  // 1) embeddings (f16, padded to 224)
  embed_kernel<<<BT, 64, 0, stream>>>(tau, oi, di, ti, wi,
                                      Wo, bo, Wd, bd, Wt, bt, Ww, bw, xp);
  // 2) QKV projections, z = head; q/k/v layout [H][B*T][32] f16
  dim3 gq(1, 128, 8);   // NT=2 -> 2 tiles cover N=32
  gemm_wmma<2><<<gq, 32, 0, stream>>>(xp, Wqt, bq, nullptr, qbuf, 32, FINP,
                                      0L, (long)32 * FINP, 32L, (long)BT * 32, 0);
  gemm_wmma<2><<<gq, 32, 0, stream>>>(xp, Wkt, bk, nullptr, kbuf, 32, FINP,
                                      0L, (long)32 * FINP, 32L, (long)BT * 32, 0);
  gemm_wmma<2><<<gq, 32, 0, stream>>>(xp, Wvt, bv, nullptr, vbuf, 32, FINP,
                                      0L, (long)32 * FINP, 32L, (long)BT * 32, 0);
  // 3) causal attention (f16 in/out)
  attn_kernel<<<16 * HEADS, 256, 0, stream>>>(qbuf, kbuf, vbuf, attnx);
  // 4) hidden = gelu(attn @ Watt) -> f32 output 0 + f16 working copy
  gemm_wmma<4><<<dim3(4, 128, 1), 32, 0, stream>>>(attnx, Wattt, nullptr,
                                                   hidden, hiddenh, 256, 256,
                                                   0, 0, 0, 0, 1);
  // 5) raw = hidden @ Wtm + btm (f32 for softmax/softplus math)
  gemm_wmma<2><<<dim3(1, 128, 1), 32, 0, stream>>>(hiddenh, Wtmt, btm,
                                                   rawb, nullptr, 32, 256,
                                                   0, 0, 0, 0, 0);
  // 6) time-mix params + f16 o_hidden
  timemix_kernel<<<BT, 64, 0, stream>>>(rawb, hiddenh, ohidh);
  // 7) o_out = softmax(o_hidden @ Wol + bol) -> f32 output 1 (logits in-place)
  gemm_wmma<4><<<dim3(4, 128, 1), 32, 0, stream>>>(ohidh, Wolt, bol,
                                                   oout, nullptr, 256, OHID,
                                                   0, 0, 0, 0, 0);
  softmax256_kernel<<<BT, 256, 0, stream>>>(oout);
  // 8) d1/d2 low-rank factors -> f16 (only consumer wants f16)
  gemm_wmma<4><<<dim3(64, 128, 1), 32, 0, stream>>>(ohidh, Wd1t, bd1,
                                                    nullptr, d1b, 4096, OHID,
                                                    0, 0, 0, 0, 0);
  gemm_wmma<4><<<dim3(64, 128, 1), 32, 0, stream>>>(ohidh, Wd2t, bd2,
                                                    nullptr, d2b, 4096, OHID,
                                                    0, 0, 0, 0, 0);
  // 9) fused od -> mask -> softmax -> weighted sum -> f32 output 2
  od_kernel<<<BT, 256, 0, stream>>>(d1b, d2b, oout, dfin);
}
